// HoughTransformerEncoderLayer_9320079032802
// MI455X (gfx1250) — matrix operand order repaired and checked by reading, not compiled, not run on hardware
//
#include <hip/hip_runtime.h>
#include <hip/hip_bf16.h>

// ---------------- problem constants ----------------
#define B_    2
#define NTOK  13294            // 100*100 + 50*50 + 25*25 + 13*13
#define E_    256
#define NH_   8
#define DH_   32
#define NL_   4
#define NP__  4
#define KSEL  300
#define NC_   91
#define DFF_  1024
#define BN_   (B_ * NTOK)

typedef __bf16 bf16_t;
typedef __attribute__((ext_vector_type(16))) __bf16 v16bf;
typedef __attribute__((ext_vector_type(8)))  float  v8f;

// =====================================================================
// Generic bf16 WMMA GEMM:  C[M,N] = A[M,K] @ W[N,K]^T + bias
// A row-major bf16 (lda=K), W row-major bf16 (N rows of K) -> B tile is W^T.
// 128 threads = 4 waves; block tile 64x64; wave tile 32x32; K step 32.
// LDS tiles are filled with gfx1250 async global->LDS DMA (ASYNCcnt) on the
// steady-state path; the M-tail tile falls back to masked loads + ds_store.
// =====================================================================
template <bool RELU>
__global__ __launch_bounds__(128)
void gemm_bf16_wmma(const bf16_t* __restrict__ A, const bf16_t* __restrict__ W,
                    const float* __restrict__ bias,
                    float* __restrict__ outF, bf16_t* __restrict__ outB,
                    int M, int N, int K) {
    __shared__ bf16_t sA[64][32];
    __shared__ bf16_t sB[64][32];

    const int tid  = threadIdx.x;
    const int wave = tid >> 5;
    const int lane = tid & 31;
    const int waveM = (wave >> 1) * 32;   // 0 / 32
    const int waveN = (wave & 1) * 32;    // 0 / 32
    const int blockRow = blockIdx.y * 64;
    const int blockCol = blockIdx.x * 64;

    v8f acc[2][2];
#pragma unroll
    for (int i = 0; i < 2; ++i)
#pragma unroll
        for (int j = 0; j < 2; ++j)
#pragma unroll
            for (int r = 0; r < 8; ++r) acc[i][j][r] = 0.0f;

    const int lrow = tid >> 1;          // 0..63
    const int lcol = (tid & 1) * 16;    // 0 / 16 (16 bf16 = 32B per thread)

    const int h  = lane >> 4;
    const int ml = lane & 15;

    const bool fullA = (blockRow + 64) <= M;   // tail tiles need zero-fill path

    // Per-lane LDS byte offsets (flat shared pointer low 32 bits == LDS offset)
    const unsigned ldsA = (unsigned)(uintptr_t)&sA[lrow][lcol];
    const unsigned ldsB = (unsigned)(uintptr_t)&sB[lrow][lcol];
    // Per-lane global byte addresses for this thread's 32B row chunk
    const unsigned long long gA =
        (unsigned long long)(uintptr_t)(A + (size_t)min(blockRow + lrow, M - 1) * K + lcol);
    const unsigned long long gB =
        (unsigned long long)(uintptr_t)(W + (size_t)(blockCol + lrow) * K + lcol);

    for (int k0 = 0; k0 < K; k0 += 32) {
        const unsigned long long kb = (unsigned long long)k0 * 2;  // bytes

        // ---- B tile: always full (all our N are multiples of 64) -> async DMA
        {
            unsigned long long a0 = gB + kb;
            asm volatile("global_load_async_to_lds_b128 %0, %1, off"
                         :: "v"(ldsB), "v"(a0) : "memory");
            asm volatile("global_load_async_to_lds_b128 %0, %1, off offset:16"
                         :: "v"(ldsB), "v"(a0) : "memory");
        }
        // ---- A tile ----
        if (fullA) {
            unsigned long long a0 = gA + kb;
            asm volatile("global_load_async_to_lds_b128 %0, %1, off"
                         :: "v"(ldsA), "v"(a0) : "memory");
            asm volatile("global_load_async_to_lds_b128 %0, %1, off offset:16"
                         :: "v"(ldsA), "v"(a0) : "memory");
        } else {
            uint4 z0 = make_uint4(0u, 0u, 0u, 0u), z1 = z0;
            int gr = blockRow + lrow;
            if (gr < M) {
                const uint4* src =
                    reinterpret_cast<const uint4*>(A + (size_t)gr * K + k0 + lcol);
                z0 = src[0]; z1 = src[1];
            }
            uint4* dst = reinterpret_cast<uint4*>(&sA[lrow][lcol]);
            dst[0] = z0; dst[1] = z1;
        }

        // prefetch next K panel while DMA completes
        if (k0 + 32 < K) {
            __builtin_prefetch((const void*)(gA + kb + 64), 0, 1);
            __builtin_prefetch((const void*)(gB + kb + 64), 0, 1);
        }

        // wait for this wave's async DMA into LDS, then workgroup barrier
        asm volatile("s_wait_asynccnt 0x0" ::: "memory");
        __syncthreads();

        union Frag { v16bf v; bf16_t e[16]; };
        Frag af[2], bf[2];
#pragma unroll
        for (int i = 0; i < 2; ++i) {
            // A frag: lane m=ml holds K = {8h..8h+7} and {16+8h..23+8h}
            const bf16_t* pa = &sA[waveM + i * 16 + ml][0];
#pragma unroll
            for (int j = 0; j < 8; ++j) {
                af[i].e[j]     = pa[8 * h + j];
                af[i].e[8 + j] = pa[16 + 8 * h + j];
            }
            // B frag: lane n=ml holds K = {8h..8h+7} and {16+8h..23+8h}
            const bf16_t* pb = &sB[waveN + i * 16 + ml][0];
#pragma unroll
            for (int j = 0; j < 8; ++j) {
                bf[i].e[j]     = pb[8 * h + j];
                bf[i].e[8 + j] = pb[16 + 8 * h + j];
            }
        }
#pragma unroll
        for (int i = 0; i < 2; ++i)
#pragma unroll
            for (int j = 0; j < 2; ++j)
                acc[i][j] = __builtin_amdgcn_wmma_f32_16x16x32_bf16(
                    false, af[i].v, false, bf[j].v, (short)0, acc[i][j],
                    false, false);
        __syncthreads();
    }

    // ---- epilogue: C element r <-> (M = r + 8h, N = ml) per 16x16 tile ----
#pragma unroll
    for (int i = 0; i < 2; ++i)
#pragma unroll
        for (int j = 0; j < 2; ++j) {
            int col = blockCol + waveN + j * 16 + ml;
            float bv = bias ? bias[col] : 0.0f;
#pragma unroll
            for (int r = 0; r < 8; ++r) {
                int row = blockRow + waveM + i * 16 + r + 8 * h;
                if (row < M && col < N) {
                    float v = acc[i][j][r] + bv;
                    if (RELU) v = fmaxf(v, 0.0f);
                    size_t o = (size_t)row * N + col;
                    if (outF) outF[o] = v;
                    if (outB) outB[o] = (bf16_t)v;
                }
            }
        }
}

// =====================================================================
// Small helper kernels
// =====================================================================
__global__ void cvt_f32_bf16_kernel(const float* __restrict__ in,
                                    bf16_t* __restrict__ out, int n) {
    int i = blockIdx.x * blockDim.x + threadIdx.x;
    if (i < n) out[i] = (bf16_t)in[i];
}

__global__ void mc_kernel(const float* __restrict__ score,
                          const float* __restrict__ fg,
                          float* __restrict__ mc) {
    int i = blockIdx.x * blockDim.x + threadIdx.x;
    if (i >= BN_) return;
    const float* s = score + (size_t)i * NC_;
    float m = s[0];
    for (int c = 1; c < NC_; ++c) m = fmaxf(m, s[c]);
    mc[i] = m * fg[i];
}

// Top-K via iterative argmax (ties -> lowest index, matches lax.top_k)
__global__ void topk_kernel(const float* __restrict__ mc,
                            float* __restrict__ scratch,
                            int* __restrict__ idx_out) {
    const int b = blockIdx.x, tid = threadIdx.x;
    for (int i = tid; i < NTOK; i += 256) scratch[b * NTOK + i] = mc[b * NTOK + i];
    __threadfence();
    __syncthreads();
    __shared__ float sv[256];
    __shared__ int   si[256];
    for (int kk = 0; kk < KSEL; ++kk) {
        float bv = -INFINITY; int bi = 0x7fffffff;
        for (int i = tid; i < NTOK; i += 256) {
            float v = scratch[b * NTOK + i];
            if (v > bv || (v == bv && i < bi)) { bv = v; bi = i; }
        }
        sv[tid] = bv; si[tid] = bi;
        __syncthreads();
        for (int s = 128; s > 0; s >>= 1) {
            if (tid < s) {
                if (sv[tid + s] > sv[tid] ||
                    (sv[tid + s] == sv[tid] && si[tid + s] < si[tid])) {
                    sv[tid] = sv[tid + s]; si[tid] = si[tid + s];
                }
            }
            __syncthreads();
        }
        if (tid == 0) {
            idx_out[b * KSEL + kk] = si[0];
            scratch[b * NTOK + si[0]] = -INFINITY;
        }
        __threadfence();
        __syncthreads();
    }
}

// gather selected tokens: sel (f32 + bf16), qk = sel + sel_pos (bf16)
__global__ void gather_kernel(const float* __restrict__ query,
                              const float* __restrict__ query_pos,
                              const int* __restrict__ idx,
                              float* __restrict__ selF, bf16_t* __restrict__ selB,
                              bf16_t* __restrict__ qkB) {
    int t = blockIdx.x * blockDim.x + threadIdx.x;
    if (t >= B_ * KSEL * E_) return;
    int e = t & 255, row = t >> 8;
    int b = row / KSEL;
    int target = idx[row];
    size_t src = ((size_t)b * NTOK + target) * E_ + e;
    float q = query[src], p = query_pos[src];
    selF[t] = q;
    selB[t] = (bf16_t)q;
    qkB[t]  = (bf16_t)(q + p);
}

// per (b,h,q) online-softmax attention over 300 keys, DH=32
__global__ void mha_attn_kernel(const float* __restrict__ qkp,
                                const float* __restrict__ vp,
                                bf16_t* __restrict__ oB) {
    int t = blockIdx.x * blockDim.x + threadIdx.x;
    if (t >= B_ * NH_ * KSEL) return;
    int q = t % KSEL, h = (t / KSEL) % NH_, b = t / (KSEL * NH_);
    const float* qrow = qkp + ((size_t)(b * KSEL + q)) * 512 + h * DH_;
    const float scale = 0.17677669529663687f; // 1/sqrt(32)
    float m = -INFINITY, l = 0.0f, acc[DH_];
#pragma unroll
    for (int d = 0; d < DH_; ++d) acc[d] = 0.0f;
    for (int j = 0; j < KSEL; ++j) {
        const float* krow = qkp + ((size_t)(b * KSEL + j)) * 512 + 256 + h * DH_;
        float s = 0.0f;
#pragma unroll
        for (int d = 0; d < DH_; ++d) s += qrow[d] * krow[d];
        s *= scale;
        float mn   = fmaxf(m, s);
        float corr = __expf(m - mn);
        float w    = __expf(s - mn);
        l = l * corr + w;
        const float* vrow = vp + ((size_t)(b * KSEL + j)) * E_ + h * DH_;
#pragma unroll
        for (int d = 0; d < DH_; ++d) acc[d] = acc[d] * corr + w * vrow[d];
        m = mn;
    }
    float inv = 1.0f / l;
    bf16_t* orow = oB + ((size_t)(b * KSEL + q)) * E_ + h * DH_;
#pragma unroll
    for (int d = 0; d < DH_; ++d) orow[d] = (bf16_t)(acc[d] * inv);
}

// LN(sel + t2) scattered back into q_work at idx
__global__ void scatter_ln_kernel(const float* __restrict__ t2,
                                  const float* __restrict__ sel,
                                  const float* __restrict__ g,
                                  const float* __restrict__ beta,
                                  const int* __restrict__ idx,
                                  float* __restrict__ q_work) {
    int row = blockIdx.x, e = threadIdx.x;
    int b = row / KSEL;
    float v = sel[(size_t)row * E_ + e] + t2[(size_t)row * E_ + e];
    __shared__ float red[E_];
    red[e] = v; __syncthreads();
    for (int s = 128; s > 0; s >>= 1) { if (e < s) red[e] += red[e + s]; __syncthreads(); }
    float mean = red[0] / E_;
    __syncthreads();
    float d = v - mean;
    red[e] = d * d; __syncthreads();
    for (int s = 128; s > 0; s >>= 1) { if (e < s) red[e] += red[e + s]; __syncthreads(); }
    float var = red[0] / E_;
    float y = d * rsqrtf(var + 1e-5f) * g[e] + beta[e];
    int target = idx[row];
    q_work[((size_t)b * NTOK + target) * E_ + e] = y;
}

// generic residual + LN, one block per row (E=256 threads)
__global__ void ln_residual_kernel(const float* __restrict__ x,
                                   const float* __restrict__ res,
                                   const float* __restrict__ g,
                                   const float* __restrict__ beta,
                                   float* __restrict__ outF,
                                   bf16_t* __restrict__ outB) {
    int row = blockIdx.x, e = threadIdx.x;
    size_t o = (size_t)row * E_ + e;
    float v = x[o] + res[o];
    __shared__ float red[E_];
    red[e] = v; __syncthreads();
    for (int s = 128; s > 0; s >>= 1) { if (e < s) red[e] += red[e + s]; __syncthreads(); }
    float mean = red[0] / E_;
    __syncthreads();
    float d = v - mean;
    red[e] = d * d; __syncthreads();
    for (int s = 128; s > 0; s >>= 1) { if (e < s) red[e] += red[e + s]; __syncthreads(); }
    float var = red[0] / E_;
    float y = d * rsqrtf(var + 1e-5f) * g[e] + beta[e];
    if (outF) outF[o] = y;
    if (outB) outB[o] = (bf16_t)y;
}

__global__ void qpos_bf16_kernel(const float* __restrict__ q,
                                 const float* __restrict__ qp,
                                 bf16_t* __restrict__ out, int n) {
    int i = blockIdx.x * blockDim.x + threadIdx.x;
    if (i < n) out[i] = (bf16_t)(q[i] + qp[i]);
}

// softmax over NL*NP=16 logits per (b,n,h), in place
__global__ void softmax_aw_kernel(float* __restrict__ aw) {
    int t = blockIdx.x * blockDim.x + threadIdx.x;
    if (t >= BN_ * NH_) return;
    float* p = aw + (size_t)(t / NH_) * 128 + (t % NH_) * 16;
    float m = p[0];
#pragma unroll
    for (int i = 1; i < 16; ++i) m = fmaxf(m, p[i]);
    float s = 0.0f;
#pragma unroll
    for (int i = 0; i < 16; ++i) { float e = __expf(p[i] - m); p[i] = e; s += e; }
    float inv = 1.0f / s;
#pragma unroll
    for (int i = 0; i < 16; ++i) p[i] *= inv;
}

// multi-scale deformable sampling; thread = (b,n,h,dim-group of 8)
__global__ void msda_kernel(const float* __restrict__ vh,
                            const float* __restrict__ off,
                            const float* __restrict__ aw,
                            const float* __restrict__ refp,
                            bf16_t* __restrict__ outB) {
    int t = blockIdx.x * blockDim.x + threadIdx.x;
    if (t >= BN_ * NH_ * 4) return;
    int gg = t & 3, h = (t >> 2) & 7;
    int bn = t >> 5;
    int b = bn / NTOK;
    const int WV[NL_] = {100, 50, 25, 13};
    const int ST[NL_] = {0, 10000, 12500, 13125};
    const float* offrow = off + (size_t)bn * 256;
    const float* awrow  = aw  + (size_t)bn * 128;
    float acc[8];
#pragma unroll
    for (int d = 0; d < 8; ++d) acc[d] = 0.0f;
    for (int lvl = 0; lvl < NL_; ++lvl) {
        const int Wl = WV[lvl], Hl = WV[lvl], st = ST[lvl];
        float rx = refp[((size_t)bn * NL_ + lvl) * 2 + 0];
        float ry = refp[((size_t)bn * NL_ + lvl) * 2 + 1];
        for (int p = 0; p < NP__; ++p) {
            float ox = offrow[h * 32 + lvl * 8 + p * 2 + 0];
            float oy = offrow[h * 32 + lvl * 8 + p * 2 + 1];
            float px = (rx + ox / (float)Wl) * Wl - 0.5f;
            float py = (ry + oy / (float)Hl) * Hl - 0.5f;
            float x0f = floorf(px), y0f = floorf(py);
            float fx = px - x0f, fy = py - y0f;
            int x0 = (int)x0f, y0 = (int)y0f;
            float wgt = awrow[h * 16 + lvl * 4 + p];
            float cw[4] = {(1 - fx) * (1 - fy) * wgt, fx * (1 - fy) * wgt,
                           (1 - fx) * fy * wgt,       fx * fy * wgt};
            int cx[4] = {x0, x0 + 1, x0, x0 + 1};
            int cy[4] = {y0, y0, y0 + 1, y0 + 1};
#pragma unroll
            for (int c = 0; c < 4; ++c) {
                int xi = cx[c], yi = cy[c];
                if (xi >= 0 && xi < Wl && yi >= 0 && yi < Hl) {
                    const float* vp = vh +
                        (((size_t)b * NTOK + st + yi * Wl + xi) * NH_ + h) * DH_ + gg * 8;
                    float w = cw[c];
#pragma unroll
                    for (int d = 0; d < 8; ++d) acc[d] += w * vp[d];
                }
            }
        }
    }
    bf16_t* orow = outB + ((size_t)bn * NH_ + h) * DH_ + gg * 8;
#pragma unroll
    for (int d = 0; d < 8; ++d) orow[d] = (bf16_t)acc[d];
}

// =====================================================================
// host side
// =====================================================================
static inline void* bump(char*& p, size_t bytes) {
    void* r = (void*)p;
    p += (bytes + 255) & ~(size_t)255;
    return r;
}

static inline void run_gemm(const bf16_t* A, const bf16_t* W, const float* bias,
                            float* outF, bf16_t* outB, int M, int N, int K,
                            bool relu, hipStream_t stream) {
    dim3 grid(N / 64, (M + 63) / 64);
    if (relu)
        gemm_bf16_wmma<true><<<grid, 128, 0, stream>>>(A, W, bias, outF, outB, M, N, K);
    else
        gemm_bf16_wmma<false><<<grid, 128, 0, stream>>>(A, W, bias, outF, outB, M, N, K);
}

extern "C" void kernel_launch(void* const* d_in, const int* in_sizes, int n_in,
                              void* d_out, int out_size, void* d_ws, size_t ws_size,
                              hipStream_t stream) {
    (void)in_sizes; (void)n_in; (void)out_size; (void)ws_size;
    const float* query      = (const float*)d_in[0];
    const float* query_pos  = (const float*)d_in[1];
    const float* value      = (const float*)d_in[2];
    const float* refp       = (const float*)d_in[3];
    const float* score_tgt  = (const float*)d_in[6];
    const float* fg         = (const float*)d_in[7];
    const float* pre_in_w   = (const float*)d_in[8];
    const float* pre_in_b   = (const float*)d_in[9];
    const float* pre_out_w  = (const float*)d_in[10];
    const float* pre_out_b  = (const float*)d_in[11];
    const float* pre_norm_g = (const float*)d_in[12];
    const float* pre_norm_b = (const float*)d_in[13];
    const float* vproj_w    = (const float*)d_in[14];
    const float* vproj_b    = (const float*)d_in[15];
    const float* off_w      = (const float*)d_in[16];
    const float* off_b      = (const float*)d_in[17];
    const float* aw_w       = (const float*)d_in[18];
    const float* aw_b       = (const float*)d_in[19];
    const float* oproj_w    = (const float*)d_in[20];
    const float* oproj_b    = (const float*)d_in[21];
    const float* norm1_g    = (const float*)d_in[22];
    const float* norm1_b    = (const float*)d_in[23];
    const float* lin1_w     = (const float*)d_in[24];
    const float* lin1_b     = (const float*)d_in[25];
    const float* lin2_w     = (const float*)d_in[26];
    const float* lin2_b     = (const float*)d_in[27];
    const float* norm2_g    = (const float*)d_in[28];
    const float* norm2_b    = (const float*)d_in[29];

    char* p = (char*)d_ws;
    const size_t NE = (size_t)BN_ * E_;

    float*  mc      = (float*)bump(p, BN_ * 4);
    float*  mc_scr  = (float*)bump(p, BN_ * 4);
    int*    idx     = (int*)  bump(p, B_ * KSEL * 4);
    float*  selF    = (float*)bump(p, B_ * KSEL * E_ * 4);
    bf16_t* selB    = (bf16_t*)bump(p, B_ * KSEL * E_ * 2);
    bf16_t* qkB     = (bf16_t*)bump(p, B_ * KSEL * E_ * 2);
    float*  qkp     = (float*)bump(p, B_ * KSEL * 512 * 4);
    float*  vp      = (float*)bump(p, B_ * KSEL * E_ * 4);
    bf16_t* oB      = (bf16_t*)bump(p, B_ * KSEL * E_ * 2);
    float*  t2      = (float*)bump(p, B_ * KSEL * E_ * 4);
    // bf16 weights
    bf16_t* w_in    = (bf16_t*)bump(p, 768 * 256 * 2);
    bf16_t* w_po    = (bf16_t*)bump(p, 256 * 256 * 2);
    bf16_t* w_vp    = (bf16_t*)bump(p, 256 * 256 * 2);
    bf16_t* w_off   = (bf16_t*)bump(p, 256 * 256 * 2);
    bf16_t* w_aw    = (bf16_t*)bump(p, 128 * 256 * 2);
    bf16_t* w_op    = (bf16_t*)bump(p, 256 * 256 * 2);
    bf16_t* w_l1    = (bf16_t*)bump(p, (size_t)DFF_ * 256 * 2);
    bf16_t* w_l2    = (bf16_t*)bump(p, (size_t)256 * DFF_ * 2);
    // big activations
    float*  q_work  = (float*)bump(p, NE * 4);
    bf16_t* qposB   = (bf16_t*)bump(p, NE * 2);
    bf16_t* valueB  = (bf16_t*)bump(p, NE * 2);
    float*  vh      = (float*)bump(p, NE * 4);
    float*  offbuf  = (float*)bump(p, NE * 4);
    float*  awbuf   = (float*)bump(p, (size_t)BN_ * 128 * 4);
    bf16_t* msdaB   = (bf16_t*)bump(p, NE * 2);
    float*  src2    = (float*)bump(p, NE * 4);
    float*  q1F     = (float*)bump(p, NE * 4);
    bf16_t* q1B     = (bf16_t*)bump(p, NE * 2);
    bf16_t* ff1B    = (bf16_t*)bump(p, (size_t)BN_ * DFF_ * 2);
    float*  ff2     = (float*)bump(p, NE * 4);

    const int T = 256;
    auto blks = [](size_t n, int t) { return (int)((n + t - 1) / t); };

    // weight + value conversion to bf16
    cvt_f32_bf16_kernel<<<blks(768 * 256, T), T, 0, stream>>>(pre_in_w, w_in, 768 * 256);
    cvt_f32_bf16_kernel<<<blks(256 * 256, T), T, 0, stream>>>(pre_out_w, w_po, 256 * 256);
    cvt_f32_bf16_kernel<<<blks(256 * 256, T), T, 0, stream>>>(vproj_w, w_vp, 256 * 256);
    cvt_f32_bf16_kernel<<<blks(256 * 256, T), T, 0, stream>>>(off_w, w_off, 256 * 256);
    cvt_f32_bf16_kernel<<<blks(128 * 256, T), T, 0, stream>>>(aw_w, w_aw, 128 * 256);
    cvt_f32_bf16_kernel<<<blks(256 * 256, T), T, 0, stream>>>(oproj_w, w_op, 256 * 256);
    cvt_f32_bf16_kernel<<<blks((size_t)DFF_ * 256, T), T, 0, stream>>>(lin1_w, w_l1, DFF_ * 256);
    cvt_f32_bf16_kernel<<<blks((size_t)DFF_ * 256, T), T, 0, stream>>>(lin2_w, w_l2, DFF_ * 256);
    cvt_f32_bf16_kernel<<<blks(NE, T), T, 0, stream>>>(value, valueB, (int)NE);

    // top-K selection
    mc_kernel<<<blks(BN_, T), T, 0, stream>>>(score_tgt, fg, mc);
    topk_kernel<<<B_, 256, 0, stream>>>(mc, mc_scr, idx);
    gather_kernel<<<blks(B_ * KSEL * E_, T), T, 0, stream>>>(query, query_pos, idx,
                                                             selF, selB, qkB);

    // MHA over selected tokens (WMMA projections, scalar flash core)
    run_gemm(qkB, w_in, pre_in_b, qkp, nullptr, B_ * KSEL, 512, 256, false, stream);
    run_gemm(selB, w_in + (size_t)512 * 256, pre_in_b + 512, vp, nullptr,
             B_ * KSEL, 256, 256, false, stream);
    mha_attn_kernel<<<blks(B_ * NH_ * KSEL, 128), 128, 0, stream>>>(qkp, vp, oB);
    run_gemm(oB, w_po, pre_out_b, t2, nullptr, B_ * KSEL, 256, 256, false, stream);

    // scatter LN(sel+t2) back into a working copy of query
    hipMemcpyAsync(q_work, query, NE * 4, hipMemcpyDeviceToDevice, stream);
    scatter_ln_kernel<<<B_ * KSEL, E_, 0, stream>>>(t2, selF, pre_norm_g, pre_norm_b,
                                                    idx, q_work);

    // q_pos (bf16) and the three big projections
    qpos_bf16_kernel<<<blks(NE, T), T, 0, stream>>>(q_work, query_pos, qposB, (int)NE);
    run_gemm(valueB, w_vp, vproj_b, vh, nullptr, BN_, 256, 256, false, stream);
    run_gemm(qposB, w_off, off_b, offbuf, nullptr, BN_, 256, 256, false, stream);
    run_gemm(qposB, w_aw, aw_b, awbuf, nullptr, BN_, 128, 256, false, stream);
    softmax_aw_kernel<<<blks((size_t)BN_ * NH_, T), T, 0, stream>>>(awbuf);

    // deformable sampling + output projection
    msda_kernel<<<blks((size_t)BN_ * NH_ * 4, T), T, 0, stream>>>(vh, offbuf, awbuf,
                                                                  refp, msdaB);
    run_gemm(msdaB, w_op, oproj_b, src2, nullptr, BN_, 256, 256, false, stream);
    ln_residual_kernel<<<BN_, E_, 0, stream>>>(src2, q_work, norm1_g, norm1_b, q1F, q1B);

    // FFN
    run_gemm(q1B, w_l1, lin1_b, nullptr, ff1B, BN_, DFF_, 256, true, stream);
    run_gemm(ff1B, w_l2, lin2_b, ff2, nullptr, BN_, 256, DFF_, false, stream);
    ln_residual_kernel<<<BN_, E_, 0, stream>>>(ff2, q1F, norm2_g, norm2_b,
                                               (float*)d_out, nullptr);
}
